// Attention_51470888075468
// MI455X (gfx1250) — compile-verified, matrix operand order checked by reading, not run on hardware
//
#include <hip/hip_runtime.h>
#include <math.h>

// ---------------------------------------------------------------------------
// PVT-style spatial-reduction attention for MI455X (gfx1250, wave32, WMMA).
// bf16 storage for all activations/weights, f32 WMMA accumulation.
// ---------------------------------------------------------------------------

typedef __attribute__((ext_vector_type(16))) __bf16 v16bf;
typedef __attribute__((ext_vector_type(8)))  __bf16 v8bf;
typedef __attribute__((ext_vector_type(8)))  float  v8f;

#define WMMA_BF16(A, B, Cacc) \
  __builtin_amdgcn_wmma_f32_16x16x32_bf16(false, (A), false, (B), (short)0, (Cacc), false, false)

static constexpr int kB   = 32;     // batch
static constexpr int kN   = 1280;   // tokens (1024 search + 256 template)
static constexpr int kC   = 256;    // channels
static constexpr int kNH  = 8;      // heads
static constexpr int kHD  = 32;     // head dim
static constexpr int kM   = 320;    // reduced tokens (256 + 64)
static constexpr float kScale = 0.17677669529663687f;  // 1/sqrt(32)

// ---- fragment loader (bf16 source) ----------------------------------------
// A-fragment (16x32): lane&15 = row M, kb = (lane>>4)*8; per lane two
// contiguous 16B halves (K = kb..kb+7 and kb+16..kb+23) -> 2x b128 loads.
// Also serves as the B-fragment of D = A*W^T (column n of W^T = row n of W),
// and as the B-fragment of O = P*V when V is stored transposed [dim][key].
__device__ __forceinline__ v16bf frag_rows_h(const __bf16* p0, size_t ld, unsigned lane) {
  const __bf16* p = p0 + (size_t)(lane & 15u) * ld + (lane >> 4) * 8u;
  const v8bf lo = *(const v8bf*)p;
  const v8bf hi = *(const v8bf*)(p + 16);
  return __builtin_shufflevector(lo, hi, 0, 1, 2, 3, 4, 5, 6, 7,
                                 8, 9, 10, 11, 12, 13, 14, 15);
}

// ---- casts / repacks -------------------------------------------------------
__global__ __launch_bounds__(256) void cast_f32_bf16(
    const float* __restrict__ src, __bf16* __restrict__ dst, int n)
{
  const int i = (int)(blockIdx.x * blockDim.x + threadIdx.x) * 4;
  if (i + 3 < n) {
#pragma unroll
    for (int e = 0; e < 4; ++e) dst[i + e] = (__bf16)src[i + e];
  } else {
    for (int e = 0; e < 4 && i + e < n; ++e) dst[i + e] = (__bf16)src[i + e];
  }
}

// sr_w [co][ci][ky][kx] -> bf16 [co][kk][ci]  (kk = ky*2+kx), 256*1024 elems
__global__ __launch_bounds__(256) void repack_srw(
    const float* __restrict__ src, __bf16* __restrict__ dst)
{
  const int i = (int)(blockIdx.x * blockDim.x + threadIdx.x);   // 262144 total
  const int co = i >> 10, r = i & 1023, ci = r >> 2, kk = r & 3;
  dst[co * 1024 + kk * 256 + ci] = (__bf16)src[i];
}

// vT[b][h][d][m] = kv[b][m][256 + h*32 + d]  (coalesced writes along m)
__global__ __launch_bounds__(256) void transpose_v(
    const __bf16* __restrict__ kv, __bf16* __restrict__ vT)
{
  const int i = (int)(blockIdx.x * blockDim.x + threadIdx.x);   // B*H*32*320
  const int m = i % kM;
  const int rest = i / kM;
  const int d = rest & 31;
  const int h = (rest >> 5) & 7;
  const int b = rest >> 8;
  vT[i] = kv[((size_t)b * kM + m) * (2 * kC) + kC + h * kHD + d];
}

// ---- D = alpha * (A * W^T) + bias, wave computes 16x64 (4 accumulators) ----
template <typename TOUT>
__global__ __launch_bounds__(256) void gemm_h(
    const __bf16* __restrict__ A, const __bf16* __restrict__ W,
    const float* __restrict__ bias, TOUT* __restrict__ D,
    int M, int N, int K, float alpha)
{
  const unsigned lane = threadIdx.x & 31u;
  const int wid = (int)((blockIdx.x * blockDim.x + threadIdx.x) >> 5);
  const int tiles_n = N >> 6;                       // 64-wide column tiles
  const int total = (M >> 4) * tiles_n;
  const int nw = (int)((gridDim.x * blockDim.x) >> 5);
  for (int t = wid; t < total; t += nw) {
    const int tm = t / tiles_n, tn = t - tm * tiles_n;
    const __bf16* Ab = A + (size_t)tm * 16 * K;
    const __bf16* Wb = W + (size_t)tn * 64 * K;
    v8f acc[4] = {{}, {}, {}, {}};
    for (int k0 = 0; k0 < K; k0 += 32) {
      const v16bf a = frag_rows_h(Ab + k0, K, lane);   // reused 4x
#pragma unroll
      for (int j = 0; j < 4; ++j) {
        const v16bf b = frag_rows_h(Wb + (size_t)j * 16 * K + k0, K, lane);
        acc[j] = WMMA_BF16(a, b, acc[j]);
      }
    }
    const unsigned col = lane & 15u, mb = (lane >> 4) * 8u;
    TOUT* Db = D + (size_t)(tm * 16) * N + tn * 64;
#pragma unroll
    for (int j = 0; j < 4; ++j) {
      const float bv = bias ? bias[tn * 64 + j * 16 + col] : 0.0f;
#pragma unroll
      for (int r = 0; r < 8; ++r)
        Db[(size_t)(mb + r) * N + j * 16 + col] = (TOUT)(acc[j][r] * alpha + bv);
    }
  }
}

// ---- strided 2x2 conv as gathered GEMM (K = 4*C ordered (kk,ci)) -----------
__device__ __forceinline__ int conv_src_tok(int m, int kk) {
  const int ky = kk >> 1, kx = kk & 1;
  if (m < 256) {            // search branch: 32x32 -> 16x16
    const int oy = m >> 4, ox = m & 15;
    return (2 * oy + ky) * 32 + (2 * ox + kx);
  }
  const int tt = m - 256;   // template branch: 16x16 -> 8x8
  const int oy = tt >> 3, ox = tt & 7;
  return 1024 + (2 * oy + ky) * 16 + (2 * ox + kx);
}

__global__ __launch_bounds__(256) void srconv_h(
    const __bf16* __restrict__ xh, const __bf16* __restrict__ w2,
    const float* __restrict__ srb, float* __restrict__ y)
{
  const unsigned lane = threadIdx.x & 31u;
  const int wid = (int)((blockIdx.x * blockDim.x + threadIdx.x) >> 5);
  const int total = kB * (kM / 16) * (kC / 64);     // 32*20*4 wave-tiles
  const int nw = (int)((gridDim.x * blockDim.x) >> 5);
  for (int t = wid; t < total; t += nw) {
    const int b = t / 80;
    const int rem = t - b * 80;
    const int tm = rem >> 2, tn4 = rem & 3;
    const int m = tm * 16 + (int)(lane & 15u);
    const unsigned kb = (lane >> 4) * 8u;
    v8f acc[4] = {{}, {}, {}, {}};
#pragma unroll
    for (int kk = 0; kk < 4; ++kk) {
      const int tok = conv_src_tok(m, kk);
      const __bf16* ap = xh + ((size_t)b * kN + tok) * kC;     // per-lane row
      for (int ci0 = 0; ci0 < kC; ci0 += 32) {
        const __bf16* pa = ap + ci0 + kb;
        const v8bf alo = *(const v8bf*)pa;
        const v8bf ahi = *(const v8bf*)(pa + 16);
        const v16bf a = __builtin_shufflevector(alo, ahi, 0, 1, 2, 3, 4, 5, 6, 7,
                                                8, 9, 10, 11, 12, 13, 14, 15);
#pragma unroll
        for (int j = 0; j < 4; ++j) {
          const unsigned co = (unsigned)tn4 * 64u + j * 16u + (lane & 15u);
          const __bf16* pw = w2 + (size_t)co * 1024 + kk * 256 + ci0 + kb;
          const v8bf wlo = *(const v8bf*)pw;
          const v8bf whi = *(const v8bf*)(pw + 16);
          const v16bf w = __builtin_shufflevector(wlo, whi, 0, 1, 2, 3, 4, 5, 6, 7,
                                                  8, 9, 10, 11, 12, 13, 14, 15);
          acc[j] = WMMA_BF16(a, w, acc[j]);
        }
      }
    }
    const unsigned col = lane & 15u, mb = (lane >> 4) * 8u;
    float* yb = y + ((size_t)b * kM + tm * 16) * kC + tn4 * 64;
#pragma unroll
    for (int j = 0; j < 4; ++j) {
      const float bv = srb[tn4 * 64 + j * 16 + col];
#pragma unroll
      for (int r = 0; r < 8; ++r)
        yb[(size_t)(mb + r) * kC + j * 16 + col] = acc[j][r] + bv;
    }
  }
}

// ---- LayerNorm over C=256: f32 in -> bf16 out ------------------------------
__global__ __launch_bounds__(256) void ln256(
    const float* __restrict__ y, const float* __restrict__ g,
    const float* __restrict__ b, __bf16* __restrict__ yh)
{
  __shared__ float red[256];
  const int row = blockIdx.x;
  const int tx = threadIdx.x;
  const float v = y[(size_t)row * 256 + tx];
  red[tx] = v; __syncthreads();
#pragma unroll
  for (int s = 128; s > 0; s >>= 1) { if (tx < s) red[tx] += red[tx + s]; __syncthreads(); }
  const float mu = red[0] * (1.0f / 256.0f);
  __syncthreads();
  const float d = v - mu;
  red[tx] = d * d; __syncthreads();
#pragma unroll
  for (int s = 128; s > 0; s >>= 1) { if (tx < s) red[tx] += red[tx + s]; __syncthreads(); }
  const float var = red[0] * (1.0f / 256.0f);
  yh[(size_t)row * 256 + tx] = (__bf16)(d * rsqrtf(var + 1e-5f) * g[tx] + b[tx]);
}

// ---- flash-style attention: wave = 16 query rows of one (b,h) --------------
// qatt (bf16 [B,N,C], head h = cols h*32..+32, pre-scaled by 1/sqrt(hd))
// is overwritten in place with the attention output (private per wave).
// kv (bf16 [B,M,2C]): k = cols [0,256).  vT (bf16 [B,H,32,320]): transposed V.
__global__ __launch_bounds__(256) void attn_wmma(
    __bf16* __restrict__ qatt, const __bf16* __restrict__ kv,
    const __bf16* __restrict__ vT)
{
  __shared__ __bf16 pbuf[8][16 * 32];    // per-wave bf16 P staging (1 KB each)
  const unsigned lane = threadIdx.x & 31u;
  const unsigned wave = threadIdx.x >> 5;
  __bf16* sm = pbuf[wave];
  const int ntq = kN / 16;               // 80 row-tiles
  const int total = kB * kNH * ntq;      // 20480 wave-tiles
  const int nw = (int)((gridDim.x * blockDim.x) >> 5);
  const unsigned col = lane & 15u, hi = lane >> 4;

  for (int t = (int)(blockIdx.x * 8 + wave); t < total; t += nw) {
    const int b = t / (kNH * ntq);
    const int r0 = t - b * (kNH * ntq);
    const int h = r0 / ntq;
    const int tq = r0 - h * ntq;

    __bf16* qbase = qatt + ((size_t)b * kN + (size_t)tq * 16) * kC + h * kHD;
    const v16bf aq = frag_rows_h(qbase, kC, lane);   // q tile, K = hd = 32

    v8f o0 = {}, o1 = {};
    float mrow[8], lrow[8];
#pragma unroll
    for (int r = 0; r < 8; ++r) { mrow[r] = -1e30f; lrow[r] = 0.0f; }

    const __bf16* kvb = kv + (size_t)b * kM * (2 * kC);
    const __bf16* vtb = vT + ((size_t)(b * kNH + h)) * kHD * kM;

    for (int j = 0; j < kM; j += 32) {               // 10 key chunks of 32
      // S = q * k^T : B columns are contiguous k rows (2x b128 per frag)
      const v16bf bk0 = frag_rows_h(kvb + (size_t)j * 512 + h * kHD, 512, lane);
      const v16bf bk1 = frag_rows_h(kvb + (size_t)(j + 16) * 512 + h * kHD, 512, lane);
      const v8f z = {};
      v8f s0 = WMMA_BF16(aq, bk0, z);
      v8f s1 = WMMA_BF16(aq, bk1, z);

      float corr[8];
#pragma unroll
      for (int r = 0; r < 8; ++r) {
        float v = fmaxf(s0[r], s1[r]);
#pragma unroll
        for (int off = 1; off < 16; off <<= 1) v = fmaxf(v, __shfl_xor(v, off, 32));
        const float mn = fmaxf(mrow[r], v);
        corr[r] = __expf(mrow[r] - mn);
        mrow[r] = mn;
        s0[r] = __expf(s0[r] - mn);
        s1[r] = __expf(s1[r] - mn);
        float rs = s0[r] + s1[r];
#pragma unroll
        for (int off = 1; off < 16; off <<= 1) rs += __shfl_xor(rs, off, 32);
        lrow[r] = lrow[r] * corr[r] + rs;
        o0[r] *= corr[r];
        o1[r] *= corr[r];
      }

      // C-layout -> A-fragment relayout via per-wave bf16 LDS staging
#pragma unroll
      for (int r = 0; r < 8; ++r) {
        sm[(hi * 8 + r) * 32 + col]      = (__bf16)s0[r];
        sm[(hi * 8 + r) * 32 + 16 + col] = (__bf16)s1[r];
      }
      asm volatile("s_wait_dscnt 0x0" ::: "memory");
      v16bf pfrag;
      {
        const unsigned row = lane & 15u, kb = (lane >> 4) * 8u;
        const v8bf plo = *(const v8bf*)(sm + row * 32 + kb);        // ds_load_b128
        const v8bf phi = *(const v8bf*)(sm + row * 32 + 16 + kb);   // ds_load_b128
        pfrag = __builtin_shufflevector(plo, phi, 0, 1, 2, 3, 4, 5, 6, 7,
                                        8, 9, 10, 11, 12, 13, 14, 15);
      }

      // O += P * V : vT is [32 dims x 320 keys] -> contiguous b128 fragments
      const v16bf bv0 = frag_rows_h(vtb + j, kM, lane);
      const v16bf bv1 = frag_rows_h(vtb + (size_t)16 * kM + j, kM, lane);
      o0 = WMMA_BF16(pfrag, bv0, o0);
      o1 = WMMA_BF16(pfrag, bv1, o1);
    }

    // normalize and write back over the (private) q region
#pragma unroll
    for (int r = 0; r < 8; ++r) {
      const float inv = 1.0f / lrow[r];
      qbase[(size_t)(hi * 8 + r) * kC + col]      = (__bf16)(o0[r] * inv);
      qbase[(size_t)(hi * 8 + r) * kC + 16 + col] = (__bf16)(o1[r] * inv);
    }
  }
}

// ---------------------------------------------------------------------------
extern "C" void kernel_launch(void* const* d_in, const int* in_sizes, int n_in,
                              void* d_out, int out_size, void* d_ws, size_t ws_size,
                              hipStream_t stream)
{
  (void)in_sizes; (void)n_in; (void)out_size;
  const float* x      = (const float*)d_in[0];
  const float* q_w    = (const float*)d_in[1];
  const float* kv_w   = (const float*)d_in[2];
  const float* sr_w   = (const float*)d_in[3];
  const float* sr_b   = (const float*)d_in[4];
  const float* ln_g   = (const float*)d_in[5];
  const float* ln_b   = (const float*)d_in[6];
  const float* proj_w = (const float*)d_in[7];
  const float* proj_b = (const float*)d_in[8];

  const size_t XEL  = (size_t)kB * kN * kC;        // 10,485,760
  const size_t YEL  = (size_t)kB * kM * kC;        //  2,621,440
  const size_t KVEL = (size_t)kB * kM * 2 * kC;    //  5,242,880
  const size_t VTEL = (size_t)kB * kNH * kHD * kM; //  2,621,440
  const size_t WQ = 256 * 256, WKV = 512 * 256, WP = 256 * 256, WSR = 256 * 1024;

  char* p = (char*)d_ws;
  __bf16* xh   = (__bf16*)p;            p += XEL  * 2;   // x cast
  __bf16* qh   = (__bf16*)p;            p += XEL  * 2;   // q*scale, then attn out
  float*  yf   = (float*)p;             p += YEL  * 4;   // conv out (f32)
  __bf16* yh   = (__bf16*)p;            p += YEL  * 2;   // LN'd x_ (bf16)
  __bf16* kvh  = (__bf16*)p;            p += KVEL * 2;   // [B, M, 2C]
  __bf16* vth  = (__bf16*)p;            p += VTEL * 2;   // V transposed [B,H,32,320]
  __bf16* wqh  = (__bf16*)p;            p += WQ   * 2;
  __bf16* wkvh = (__bf16*)p;            p += WKV  * 2;
  __bf16* wph  = (__bf16*)p;            p += WP   * 2;
  __bf16* wsrh = (__bf16*)p;            p += WSR  * 2;   // repacked [co][kk][ci]
  if ((size_t)(p - (char*)d_ws) > ws_size) return;
  float* out = (float*)d_out;

  // 0) casts / repack to bf16
  cast_f32_bf16<<<(int)(XEL / 1024), 256, 0, stream>>>(x, xh, (int)XEL);
  cast_f32_bf16<<<(int)(WQ / 1024), 256, 0, stream>>>(q_w, wqh, (int)WQ);
  cast_f32_bf16<<<(int)(WKV / 1024), 256, 0, stream>>>(kv_w, wkvh, (int)WKV);
  cast_f32_bf16<<<(int)(WP / 1024), 256, 0, stream>>>(proj_w, wph, (int)WP);
  repack_srw<<<(int)(WSR / 256), 256, 0, stream>>>(sr_w, wsrh);

  // 1) q = (x @ q_w^T) * scale   [40960 x 256], K=256 -> 10240 wave-tiles
  gemm_h<__bf16><<<1280, 256, 0, stream>>>(xh, wqh, nullptr, qh,
                                           kB * kN, kC, kC, kScale);
  // 2) spatial-reduction conv (2x2 s2) as gathered GEMM, K=1024
  srconv_h<<<320, 256, 0, stream>>>(xh, wsrh, sr_b, yf);
  // 3) LayerNorm over C -> bf16
  ln256<<<kB * kM, 256, 0, stream>>>(yf, ln_g, ln_b, yh);
  // 4) kv = x_ @ kv_w^T   [10240 x 512], K=256 -> 5120 wave-tiles
  gemm_h<__bf16><<<640, 256, 0, stream>>>(yh, wkvh, nullptr, kvh,
                                          kB * kM, 2 * kC, kC, 1.0f);
  // 4b) transpose V into [B,H,32,320] for vectorized B-fragments
  transpose_v<<<(int)(VTEL / 256), 256, 0, stream>>>(kvh, vth);
  // 5) attention (flash-style), in place over qh
  attn_wmma<<<2560, 256, 0, stream>>>(qh, kvh, vth);
  // 6) out = att @ proj_w^T + proj_b  (f32 output)
  gemm_h<float><<<1280, 256, 0, stream>>>(qh, wph, proj_b, out,
                                          kB * kN, kC, kC, 1.0f);
}
